// Attention_50216757625374
// MI455X (gfx1250) — matrix operand order repaired and checked by reading, not compiled, not run on hardware
//
#include <hip/hip_runtime.h>
#include <hip/hip_bf16.h>

#define EMB 256
#define ATT 256
#define BB  8
#define SS  2048

typedef __attribute__((ext_vector_type(16))) _Float16     v16h;
typedef __attribute__((ext_vector_type(8)))  _Float16     v8h;
typedef __attribute__((ext_vector_type(8)))  float        v8f;
typedef __attribute__((ext_vector_type(4)))  unsigned int u32x4;
typedef __attribute__((ext_vector_type(8)))  int          i32x8;
typedef __attribute__((ext_vector_type(4)))  int          i32x4;

#if defined(__has_builtin)
#if __has_builtin(__builtin_amdgcn_tensor_load_to_lds)
#define HAVE_TDM 1
#endif
#endif

// ---------------------------------------------------------------------------
// Kernel 0: transpose + convert weights to f16.
// Wt layout: [3][ATT][EMB], Wt[m][a][e] = W_m[e][a]  (contraction innermost)
// ---------------------------------------------------------------------------
__global__ __launch_bounds__(256) void weights_t_kernel(
    const float* __restrict__ Wq, const float* __restrict__ Wk,
    const float* __restrict__ Wv, _Float16* __restrict__ Wt)
{
    const int idx   = blockIdx.x * 256 + threadIdx.x;
    const int which = idx >> 16;
    const int rem   = idx & 65535;
    const int a     = rem >> 8;
    const int e     = rem & 255;
    const float* W  = (which == 0) ? Wq : (which == 1) ? Wk : Wv;
    Wt[idx] = (_Float16)W[(size_t)e * ATT + a];
}

// ---------------------------------------------------------------------------
// Kernel 1: Q/K/V projections (all-vector WMMA operands).
// ---------------------------------------------------------------------------
__global__ __launch_bounds__(32) void proj_qkv_kernel(
    const float* __restrict__ x, const _Float16* __restrict__ Wt,
    const float* __restrict__ bq, const float* __restrict__ bk,
    const float* __restrict__ bv,
    _Float16* __restrict__ Qh, _Float16* __restrict__ Kh,
    _Float16* __restrict__ Vt)
{
    const int lane = threadIdx.x;
    const int l16  = lane & 15;
    const int half = lane >> 4;

    const int rowTile = blockIdx.x;
    const int colTile = blockIdx.y;
    const int which   = blockIdx.z;

    const _Float16* Wm   = Wt + (size_t)which * ATT * EMB;
    const float*    bias = (which == 0) ? bq : (which == 1) ? bk : bv;

    const int arow = rowTile * 16 + l16;
    const int col  = colTile * 16 + l16;

    v8f acc;
    const float bcol = bias[col];
#pragma unroll
    for (int r = 0; r < 8; ++r) acc[r] = bcol;

    const float*    xr = x + (size_t)arow * EMB;
    const _Float16* wr = Wm + (size_t)col * EMB;

    for (int e = 0; e < EMB; e += 32) {
        const v8f lo = *(const v8f*)(xr + e + 16 * half);
        const v8f hi = *(const v8f*)(xr + e + 16 * half + 8);
        v16h a;
#pragma unroll
        for (int j = 0; j < 8; ++j) { a[j] = (_Float16)lo[j]; a[8 + j] = (_Float16)hi[j]; }
        const v16h bfr = *(const v16h*)(wr + e + 16 * half);
        acc = __builtin_amdgcn_wmma_f32_16x16x32_f16(
            false, a, false, bfr, (short)0, acc, false, false);
    }

    if (which == 2) {
        const int bidx  = (rowTile * 16) / SS;
        const int srow0 = rowTile * 16 - bidx * SS;
        v8h vh;
#pragma unroll
        for (int r = 0; r < 8; ++r) vh[r] = (_Float16)acc[r];
        *(v8h*)(Vt + ((size_t)bidx * ATT + col) * SS + srow0 + 8 * half) = vh;
    } else {
        const float scale = (which == 0) ? 0.0625f : 1.0f;
        _Float16* Out = (which == 0) ? Qh : Kh;
#pragma unroll
        for (int r = 0; r < 8; ++r) {
            const int orow = rowTile * 16 + r + 8 * half;
            Out[(size_t)orow * ATT + col] = (_Float16)(acc[r] * scale);
        }
    }
}

#ifdef HAVE_TDM
// Build a 2-D TDM descriptor (D#) per CDNA5 ISA and issue
// TENSOR_LOAD_TO_LDS (6-arg toolchain form; groups 2/3 + extra group zero
// for <=2-D tensors).  data_size=2B, no gather/iterate/pad/multicast.
// Must be called wave-uniformly.
__device__ __forceinline__ void tdm_load_2d(
    unsigned int lds_off, const void* gptr,
    unsigned int tensor_d0, unsigned int tensor_d1,
    unsigned int tile_d0, unsigned int tile_d1,
    unsigned long long stride0_elems)
{
    const unsigned long long ga = (unsigned long long)gptr;
    u32x4 g0;
    g0[0] = 1u;                                   // count=1, user descriptor
    g0[1] = lds_off;                              // lds_addr (bytes)
    g0[2] = (unsigned int)ga;                     // global_addr[31:0]
    g0[3] = (unsigned int)((ga >> 32) & 0x1FFFFFFu) | 0x80000000u; // [56:32] | type=2
    i32x8 g1;
    g1[0] = (int)(1u << 16);                      // data_size=1 -> 2 bytes
    g1[1] = (int)((tensor_d0 & 0xFFFFu) << 16);   // tensor_dim0[15:0] @ bit48
    g1[2] = (int)(((tensor_d0 >> 16) & 0xFFFFu) | ((tensor_d1 & 0xFFFFu) << 16));
    g1[3] = (int)(((tensor_d1 >> 16) & 0xFFFFu) | ((tile_d0 & 0xFFFFu) << 16));
    g1[4] = (int)(tile_d1 & 0xFFFFu);             // tile_dim1; tile_dim2=0
    g1[5] = (int)(unsigned int)(stride0_elems & 0xFFFFFFFFull);
    g1[6] = (int)(unsigned int)((stride0_elems >> 32) & 0xFFFFull);
    g1[7] = 0;
    const i32x4 z4 = {0, 0, 0, 0};
    const i32x8 z8 = {0, 0, 0, 0, 0, 0, 0, 0};
    __builtin_amdgcn_tensor_load_to_lds(g0, g1, z4, z4, z8, 0);
}
#endif

// ---------------------------------------------------------------------------
// Kernel 2: flash attention.  8 waves/block share one batch; K/V tiles are
// staged in LDS once per block via the Tensor Data Mover (TENSORcnt +
// s_wait_tensorcnt + block barrier), cutting L2 read traffic 8x.  Each wave
// owns 16 query rows and the full 16x256 f32 accumulator; 32 WMMAs per
// 32-key iteration.
// ---------------------------------------------------------------------------
__global__ __launch_bounds__(256) void flash_attn_kernel(
    const int* __restrict__ mask,
    const _Float16* __restrict__ Qh, const _Float16* __restrict__ Kh,
    const _Float16* __restrict__ Vt, float* __restrict__ out)
{
    __shared__ __align__(16) _Float16 kbuf[32 * ATT];   // 16 KB: keys x features
    __shared__ __align__(16) _Float16 vbuf[ATT * 32];   // 16 KB: features x keys
    __shared__ __align__(16) _Float16 pbuf[8][16 * 32]; //  8 KB: per-wave P tile

    const int lane = threadIdx.x & 31;
    const int wid  = threadIdx.x >> 5;
    const int l16  = lane & 15;
    const int half = lane >> 4;

    const int tile = blockIdx.x * 8 + wid;  // 8 consecutive tiles, same batch
    const int b    = tile >> 7;
    const int qt   = tile & 127;

    const _Float16* Qb  = Qh + (size_t)b * SS * ATT;
    const _Float16* Kb  = Kh + (size_t)b * SS * ATT;
    const _Float16* Vtb = Vt + (size_t)b * ATT * SS;
    const int*      mb  = mask + (size_t)b * SS * SS;

    // ---- Q tile in registers: 8 A-fragments, two 16B loads each ----
    const int qrow = qt * 16 + l16;
    const _Float16* qr = Qb + (size_t)qrow * ATT;
    v16h qa[8];
#pragma unroll
    for (int kc = 0; kc < 8; ++kc) {
        const v8h lo = *(const v8h*)(qr + kc * 32 + 16 * half);
        const v8h hi = *(const v8h*)(qr + kc * 32 + 16 * half + 8);
#pragma unroll
        for (int j = 0; j < 8; ++j) { qa[kc][j] = lo[j]; qa[kc][8 + j] = hi[j]; }
    }

    v8f acc[16];
#pragma unroll
    for (int nc = 0; nc < 16; ++nc)
#pragma unroll
        for (int r = 0; r < 8; ++r) acc[nc][r] = 0.0f;

    float m_i[8], l_i[8];
#pragma unroll
    for (int r = 0; r < 8; ++r) { m_i[r] = -1e30f; l_i[r] = 0.0f; }

    for (int kt = 0; kt < SS; kt += 32) {
#ifdef HAVE_TDM
        __syncthreads();                 // everyone done reading previous tile
        if (wid == 0) {
            // K tile: 32 rows x 256 features, row-major into kbuf
            tdm_load_2d((unsigned int)(uintptr_t)kbuf,
                        Kb + (size_t)kt * ATT, ATT, SS, ATT, 32, ATT);
            // V tile: 256 rows x 32 keys (from transposed V), into vbuf
            tdm_load_2d((unsigned int)(uintptr_t)vbuf,
                        Vtb + kt, SS, ATT, 32, ATT, SS);
            __builtin_amdgcn_s_wait_tensorcnt(0);
        }
        __syncthreads();                 // tile visible to all waves
        const _Float16* k0 = kbuf + (size_t)l16 * ATT + 16 * half;
        const _Float16* k1 = kbuf + (size_t)(16 + l16) * ATT + 16 * half;
#else
        if (kt + 32 < SS)
            __builtin_prefetch(Kb + (size_t)(kt + 32 + l16) * ATT, 0, 1);
        const _Float16* k0 = Kb + (size_t)(kt + l16) * ATT + 16 * half;
        const _Float16* k1 = Kb + (size_t)(kt + 16 + l16) * ATT + 16 * half;
#endif

        // ---- scores: Q(16x256) @ K^T -> two 16x16 tiles ----
        v8f s0, s1;
#pragma unroll
        for (int r = 0; r < 8; ++r) { s0[r] = 0.0f; s1[r] = 0.0f; }
#pragma unroll
        for (int kc = 0; kc < 8; ++kc) {
            const v16h b0 = *(const v16h*)(k0 + kc * 32);
            const v16h b1 = *(const v16h*)(k1 + kc * 32);
            s0 = __builtin_amdgcn_wmma_f32_16x16x32_f16(
                false, qa[kc], false, b0, (short)0, s0, false, false);
            s1 = __builtin_amdgcn_wmma_f32_16x16x32_f16(
                false, qa[kc], false, b1, (short)0, s1, false, false);
        }

        // ---- mask (scale folded into Q) + rowmax ----
        float rowmax[8];
#pragma unroll
        for (int r = 0; r < 8; ++r) {
            const int mrow = qt * 16 + r + 8 * half;
            const int m0 = mb[(size_t)mrow * SS + kt + l16];
            const int m1 = mb[(size_t)mrow * SS + kt + 16 + l16];
            const float v0 = m0 ? s0[r] : -1e9f;
            const float v1 = m1 ? s1[r] : -1e9f;
            s0[r] = v0; s1[r] = v1;
            rowmax[r] = fmaxf(v0, v1);
        }
#pragma unroll
        for (int r = 0; r < 8; ++r) {
            float v = rowmax[r];
            v = fmaxf(v, __shfl_xor(v, 1, 16));
            v = fmaxf(v, __shfl_xor(v, 2, 16));
            v = fmaxf(v, __shfl_xor(v, 4, 16));
            v = fmaxf(v, __shfl_xor(v, 8, 16));
            rowmax[r] = v;
        }

        // ---- online softmax ----
        float p0[8], p1[8], alpha[8];
#pragma unroll
        for (int r = 0; r < 8; ++r) {
            const float mn = fmaxf(m_i[r], rowmax[r]);
            alpha[r] = __expf(m_i[r] - mn);
            m_i[r]   = mn;
            p0[r] = __expf(s0[r] - mn);
            p1[r] = __expf(s1[r] - mn);
            float rs = p0[r] + p1[r];
            rs += __shfl_xor(rs, 1, 16);
            rs += __shfl_xor(rs, 2, 16);
            rs += __shfl_xor(rs, 4, 16);
            rs += __shfl_xor(rs, 8, 16);
            l_i[r] = l_i[r] * alpha[r] + rs;
        }

        // ---- P tile: C-layout -> LDS -> A-layout (per-wave region) ----
        _Float16* pb = &pbuf[wid][0];
#pragma unroll
        for (int r = 0; r < 8; ++r) {
            const int prow = r + 8 * half;
            pb[prow * 32 + l16]      = (_Float16)p0[r];
            pb[prow * 32 + 16 + l16] = (_Float16)p1[r];
        }
        asm volatile("s_wait_dscnt 0" ::: "memory");   // per-wave DS in order

        v16h pa;
        {
            const _Float16* pr = pb + l16 * 32;
            const v8h lo = *(const v8h*)(pr + 16 * half);
            const v8h hi = *(const v8h*)(pr + 16 * half + 8);
#pragma unroll
            for (int j = 0; j < 8; ++j) { pa[j] = lo[j]; pa[8 + j] = hi[j]; }
        }

        // ---- rescale accumulator; acc += P(16x32) @ V(32x256) ----
#pragma unroll
        for (int nc = 0; nc < 16; ++nc)
#pragma unroll
            for (int r = 0; r < 8; ++r) acc[nc][r] *= alpha[r];

#ifdef HAVE_TDM
        const _Float16* vbase = vbuf + 16 * half;     // vbuf[f*32 + key_local]
#pragma unroll
        for (int nc = 0; nc < 16; ++nc) {
            const int fcol = nc * 16 + l16;
            const v16h vb = *(const v16h*)(vbase + (size_t)fcol * 32);
            acc[nc] = __builtin_amdgcn_wmma_f32_16x16x32_f16(
                false, pa, false, vb, (short)0, acc[nc], false, false);
        }
#else
        const _Float16* vbase = Vtb + kt + 16 * half;
#pragma unroll
        for (int nc = 0; nc < 16; ++nc) {
            const int fcol = nc * 16 + l16;
            const v16h vb = *(const v16h*)(vbase + (size_t)fcol * SS);
            acc[nc] = __builtin_amdgcn_wmma_f32_16x16x32_f16(
                false, pa, false, vb, (short)0, acc[nc], false, false);
        }
#endif
    }

    // ---- finalize ----
#pragma unroll
    for (int nc = 0; nc < 16; ++nc)
#pragma unroll
        for (int r = 0; r < 8; ++r) {
            const int orow = qt * 16 + r + 8 * half;
            out[((size_t)b * SS + orow) * ATT + nc * 16 + l16] =
                acc[nc][r] / l_i[r];
        }
}

// ---------------------------------------------------------------------------
extern "C" void kernel_launch(void* const* d_in, const int* in_sizes, int n_in,
                              void* d_out, int out_size, void* d_ws,
                              size_t ws_size, hipStream_t stream)
{
    const float* x    = (const float*)d_in[0];
    const int*   mask = (const int*)d_in[1];
    const float* Wq   = (const float*)d_in[2];
    const float* bq   = (const float*)d_in[3];
    const float* Wk   = (const float*)d_in[4];
    const float* bk   = (const float*)d_in[5];
    const float* Wv   = (const float*)d_in[6];
    const float* bv   = (const float*)d_in[7];
    float*       out  = (float*)d_out;

    const size_t wt_elems  = (size_t)3 * ATT * EMB;
    const size_t qkv_elems = (size_t)BB * SS * ATT;
    _Float16* Wt = (_Float16*)d_ws;
    _Float16* Qh = Wt + wt_elems;
    _Float16* Kh = Qh + qkv_elems;
    _Float16* Vt = Kh + qkv_elems;

    weights_t_kernel<<<dim3(3 * ATT * EMB / 256), 256, 0, stream>>>(Wq, Wk, Wv, Wt);

    dim3 gridA(BB * SS / 16, ATT / 16, 3);
    proj_qkv_kernel<<<gridA, 32, 0, stream>>>(x, Wt, bq, bk, bv, Qh, Kh, Vt);

    dim3 gridB(BB * SS / 16 / 8);
    flash_attn_kernel<<<gridB, 256, 0, stream>>>(mask, Qh, Kh, Vt, out);
}
